// Model_31903017075249
// MI455X (gfx1250) — compile-verified, matrix operand order checked by reading, not run on hardware
//
#include <hip/hip_runtime.h>
#include <hip/hip_bf16.h>
#include <cmath>

// ---------------------------------------------------------------------------
// Informer forward pass for MI455X (gfx1250, wave32, WMMA).
// All dense GEMMs run on v_wmma_f32_16x16x32_f16 (f16 in, f32 accumulate).
// Each wave computes a 32x64 output tile (2 A-frags x 4 B-frags = 8 WMMA per
// K-step) so B-fragments are reused and VMEM issue pressure is 1.5 loads/WMMA.
// ---------------------------------------------------------------------------

typedef __attribute__((ext_vector_type(16))) _Float16 v16h;
typedef __attribute__((ext_vector_type(8)))  _Float16 v8h;
typedef __attribute__((ext_vector_type(8)))  float    v8f;

#define NEG_INF (-3.4e38f)

// ---------------- WMMA GEMM -------------------------------------------------
// A: [M][K] f16 row-major.  Bt: [N][K] f16 (i.e. B transposed).  C: [M][N] f32.
// C = act(A*B + bias + res).  act: 0=none, 1=exact GELU.
// Fragment layout per CDNA5 ISA 7.12.2 (16-bit A 16x32; symmetric B 32x16):
//   lanes 0-15  : K = k0+{0..7} -> v[0..7],  K = k0+{16..23} -> v[8..15]
//   lanes 16-31 : K = k0+{8..15}-> v[0..7],  K = k0+{24..31} -> v[8..15]
__device__ inline v16h load_frag(const _Float16* base, int k1, int k2) {
    v8h lo = *(const v8h*)(base + k1);
    v8h hi = *(const v8h*)(base + k2);
    v16h r;
#pragma unroll
    for (int i = 0; i < 8; ++i) { r[i] = lo[i]; r[i + 8] = hi[i]; }
    return r;
}

__global__ __launch_bounds__(256) void k_gemm(
    const _Float16* __restrict__ A, const _Float16* __restrict__ Bt,
    const float* __restrict__ bias, const float* __restrict__ res,
    float* __restrict__ C, int M, int N, int K, int act)
{
    const int wave = threadIdx.x >> 5;
    const int lane = threadIdx.x & 31;
    const int hl   = lane >> 4;       // half-wave id
    const int m16  = lane & 15;
    const int rowBase = (blockIdx.x * 8 + wave) * 32;   // 8 waves stack in M
    const int colBase = blockIdx.y * 64;                // wave covers 64 cols
    if (rowBase >= M) return;                           // wave-uniform guard

    v8f acc[2][4] = {};
    const int kO1 = hl * 8;
    const int kO2 = 16 + hl * 8;
    const _Float16* aRow0 = A  + (size_t)(rowBase      + m16) * K;
    const _Float16* aRow1 = A  + (size_t)(rowBase + 16 + m16) * K;
    const _Float16* bp0   = Bt + (size_t)(colBase +  0 + m16) * K;
    const _Float16* bp1   = Bt + (size_t)(colBase + 16 + m16) * K;
    const _Float16* bp2   = Bt + (size_t)(colBase + 32 + m16) * K;
    const _Float16* bp3   = Bt + (size_t)(colBase + 48 + m16) * K;

    for (int k0 = 0; k0 < K; k0 += 32) {
        v16h a0 = load_frag(aRow0 + k0, kO1, kO2);
        v16h a1 = load_frag(aRow1 + k0, kO1, kO2);
        v16h f0 = load_frag(bp0 + k0, kO1, kO2);
        v16h f1 = load_frag(bp1 + k0, kO1, kO2);
        v16h f2 = load_frag(bp2 + k0, kO1, kO2);
        v16h f3 = load_frag(bp3 + k0, kO1, kO2);
        acc[0][0] = __builtin_amdgcn_wmma_f32_16x16x32_f16(false, a0, false, f0, (short)0, acc[0][0], false, false);
        acc[0][1] = __builtin_amdgcn_wmma_f32_16x16x32_f16(false, a0, false, f1, (short)0, acc[0][1], false, false);
        acc[0][2] = __builtin_amdgcn_wmma_f32_16x16x32_f16(false, a0, false, f2, (short)0, acc[0][2], false, false);
        acc[0][3] = __builtin_amdgcn_wmma_f32_16x16x32_f16(false, a0, false, f3, (short)0, acc[0][3], false, false);
        acc[1][0] = __builtin_amdgcn_wmma_f32_16x16x32_f16(false, a1, false, f0, (short)0, acc[1][0], false, false);
        acc[1][1] = __builtin_amdgcn_wmma_f32_16x16x32_f16(false, a1, false, f1, (short)0, acc[1][1], false, false);
        acc[1][2] = __builtin_amdgcn_wmma_f32_16x16x32_f16(false, a1, false, f2, (short)0, acc[1][2], false, false);
        acc[1][3] = __builtin_amdgcn_wmma_f32_16x16x32_f16(false, a1, false, f3, (short)0, acc[1][3], false, false);
    }

#pragma unroll
    for (int mt = 0; mt < 2; ++mt) {
#pragma unroll
        for (int t = 0; t < 4; ++t) {
            int col = colBase + t * 16 + m16;
            float bb = bias ? bias[col] : 0.f;
#pragma unroll
            for (int r = 0; r < 8; ++r) {
                int row = rowBase + mt * 16 + r + hl * 8;  // D: VGPR r -> M=r / M=8+r
                float v = acc[mt][t][r] + bb;
                if (res) v += res[(size_t)row * N + col];
                if (act == 1) v = 0.5f * v * (1.f + erff(v * 0.70710678f));
                C[(size_t)row * N + col] = v;
            }
        }
    }
}

// ---------------- conversion / packing kernels ------------------------------
__global__ void k_f2h(const float* __restrict__ s, _Float16* __restrict__ d, size_t n) {
    size_t id = (size_t)blockIdx.x * 256 + threadIdx.x;
    if (id < n) d[id] = (_Float16)s[id];
}

// dst[n*K+k] = (f16) src[off + n*sn + k*sk]   (weight -> transposed f16)
__global__ void k_conv_w(const float* __restrict__ src, _Float16* __restrict__ dst,
                         int N, int K, int sn, int sk, int off) {
    int id = blockIdx.x * 256 + threadIdx.x;
    if (id >= N * K) return;
    int n = id / K, k = id % K;
    dst[id] = (_Float16)src[(size_t)off + (size_t)n * sn + (size_t)k * sk];
}

// circularly shifted f16 copy of [B,L,512] activations (per batch)
__global__ void k_shift_f16(const float* __restrict__ src, _Float16* __restrict__ dst,
                            int L, int shift) {
    size_t id = (size_t)blockIdx.x * 256 + threadIdx.x;
    if (id >= (size_t)8 * L * 512) return;
    int c = (int)(id % 512); size_t row = id / 512;
    int l = (int)(row % L);  int b = (int)(row / L);
    int ls = l + shift; if (ls < 0) ls += L; if (ls >= L) ls -= L;
    dst[id] = (_Float16)src[((size_t)(b * L + ls)) * 512 + c];
}

// ---------------- elementwise / normalization -------------------------------
__global__ void k_add(const float* a, const float* b, float* c, int n) {
    int id = blockIdx.x * 256 + threadIdx.x;
    if (id < n) c[id] = a[id] + b[id];
}

// LayerNorm over D=512; optional residual (y = LN(x+res)*g + b). Block per row.
__global__ __launch_bounds__(256) void k_ln(const float* __restrict__ x,
    const float* __restrict__ res, const float* __restrict__ g,
    const float* __restrict__ b, float* __restrict__ y)
{
    __shared__ float red[256];
    int r = blockIdx.x, t = threadIdx.x;
    const float* xr = x + (size_t)r * 512;
    float v0 = xr[t], v1 = xr[t + 256];
    if (res) { v0 += res[(size_t)r * 512 + t]; v1 += res[(size_t)r * 512 + t + 256]; }
    red[t] = v0 + v1; __syncthreads();
    for (int s = 128; s > 0; s >>= 1) { if (t < s) red[t] += red[t + s]; __syncthreads(); }
    float mean = red[0] * (1.f / 512.f); __syncthreads();
    float d0 = v0 - mean, d1 = v1 - mean;
    red[t] = d0 * d0 + d1 * d1; __syncthreads();
    for (int s = 128; s > 0; s >>= 1) { if (t < s) red[t] += red[t + s]; __syncthreads(); }
    float rstd = rsqrtf(red[0] * (1.f / 512.f) + 1e-5f);
    y[(size_t)r * 512 + t]       = d0 * rstd * g[t] + b[t];
    y[(size_t)r * 512 + t + 256] = d1 * rstd * g[t + 256] + b[t + 256];
}

// ---------------- RevIN -----------------------------------------------------
__global__ void k_revin_stats(const float* __restrict__ x, float* mean, float* stdv) {
    int id = threadIdx.x; if (id >= 56) return;
    int b = id / 7, c = id % 7;
    float s = 0.f, s2 = 0.f;
    for (int l = 0; l < 720; ++l) {
        float v = x[((size_t)(b * 720 + l)) * 7 + c]; s += v; s2 += v * v;
    }
    float m = s * (1.f / 720.f);
    mean[id] = m;
    stdv[id] = sqrtf(fmaxf(s2 * (1.f / 720.f) - m * m, 0.f) + 1e-5f);
}

__global__ void k_revin_norm(const float* __restrict__ x, const float* mean,
                             const float* stdv, const float* rw, const float* rb,
                             float* __restrict__ y) {
    int id = blockIdx.x * 256 + threadIdx.x;
    if (id >= 8 * 720 * 7) return;
    int c = id % 7; int b = id / (720 * 7);
    y[id] = (x[id] - mean[b * 7 + c]) / stdv[b * 7 + c] * rw[c] + rb[c];
}

// ---------------- token embedding: circular conv(7->512,k=3) + PE -----------
__global__ __launch_bounds__(256) void k_embed(const float* __restrict__ x,
    const float* __restrict__ w, float* __restrict__ out, int L)
{
    int id = blockIdx.x * 256 + threadIdx.x;
    if (id >= 8 * L * 512) return;
    int o = id % 512; int row = id / 512; int l = row % L; int b = row / L;
    float s = 0.f;
#pragma unroll
    for (int t = 0; t < 3; ++t) {
        int ls = l + t - 1; ls = (ls < 0) ? ls + L : (ls >= L ? ls - L : ls);
        const float* xr = x + ((size_t)(b * L + ls)) * 7;
#pragma unroll
        for (int c = 0; c < 7; ++c) s += xr[c] * w[o * 21 + c * 3 + t];
    }
    float div = __expf(-(float)(o & ~1) * (9.21034037f / 512.f));   // ln(1e4)/d
    float pe  = (o & 1) ? __cosf((float)l * div) : __sinf((float)l * div);
    out[id] = s + pe;
}

// ---------------- ProbSparse attention pieces -------------------------------
__device__ inline unsigned hash_u(unsigned x) {
    x ^= x >> 16; x *= 0x7feb352du; x ^= x >> 15; x *= 0x846ca68bu; x ^= x >> 16;
    return x;
}

// sparsity measure M[b,h,l] = max_s(QK_s) - mean_over_LK.  One wave per (b,h,l).
__global__ __launch_bounds__(256) void k_sample(const float* __restrict__ qp,
    const float* __restrict__ kp, float* __restrict__ Mval,
    int Lq, int Lk, int Up, unsigned seed)
{
    int gw   = (blockIdx.x * 256 + threadIdx.x) >> 5;
    int lane = threadIdx.x & 31;
    if (gw >= 64 * Lq) return;
    int l = gw % Lq; int bh = gw / Lq; int b = bh >> 3; int h = bh & 7;
    const float* qrow = qp + ((size_t)(b * Lq + l)) * 512 + h * 64;
    float q0 = qrow[lane], q1 = qrow[lane + 32];
    float mx = NEG_INF, sm = 0.f;
    for (int s = 0; s < Up; ++s) {
        unsigned hsh = hash_u((unsigned)(l * 9781 + s * 6271) + seed * 0x9E3779B9u);
        int k = (int)(hsh % (unsigned)Lk);
        const float* krow = kp + ((size_t)(b * Lk + k)) * 512 + h * 64;
        float p = q0 * krow[lane] + q1 * krow[lane + 32];
        for (int off = 16; off; off >>= 1) p += __shfl_xor(p, off);
        mx = fmaxf(mx, p); sm += p;
    }
    if (lane == 0) Mval[gw] = mx - sm / (float)Lk;
}

// iterative top-u selection; one block per (b,h)
__global__ __launch_bounds__(256) void k_topk(const float* __restrict__ Mval,
    int* __restrict__ outIdx, int Lq, int u)
{
    __shared__ float vals[768];
    __shared__ float rv[256];
    __shared__ int   ri[256];
    __shared__ int   best;
    int t = threadIdx.x, bh = blockIdx.x;
    for (int i = t; i < Lq; i += 256) vals[i] = Mval[bh * Lq + i];
    __syncthreads();
    for (int it = 0; it < u; ++it) {
        float bv = NEG_INF; int bi = 0;
        for (int i = t; i < Lq; i += 256) if (vals[i] > bv) { bv = vals[i]; bi = i; }
        rv[t] = bv; ri[t] = bi; __syncthreads();
        for (int s = 128; s > 0; s >>= 1) {
            if (t < s && rv[t + s] > rv[t]) { rv[t] = rv[t + s]; ri[t] = ri[t + s]; }
            __syncthreads();
        }
        if (t == 0) { best = ri[0]; outIdx[bh * u + it] = best; vals[best] = NEG_INF; }
        __syncthreads();
    }
}

// full attention for selected queries; one block per (b,h,i)
__global__ __launch_bounds__(256) void k_attn_sel(const float* __restrict__ qp,
    const float* __restrict__ kp, const float* __restrict__ vp,
    const int* __restrict__ idx, float* __restrict__ upd,
    int Lq, int Lk, int u, int causal)
{
    __shared__ float sc[768];
    __shared__ float red[256];
    int t = threadIdx.x;
    int i = blockIdx.x % u; int bh = blockIdx.x / u; int b = bh >> 3; int h = bh & 7;
    int q = idx[bh * u + i];
    const float* qrow = qp + ((size_t)(b * Lq + q)) * 512 + h * 64;
    for (int k = t; k < Lk; k += 256) {
        const float* krow = kp + ((size_t)(b * Lk + k)) * 512 + h * 64;
        float d = 0.f;
#pragma unroll
        for (int dd = 0; dd < 64; ++dd) d += qrow[dd] * krow[dd];
        d *= 0.125f;                                     // 1/sqrt(64)
        if (causal && k > q) d = NEG_INF;
        sc[k] = d;
    }
    __syncthreads();
    float m = NEG_INF;
    for (int k = t; k < Lk; k += 256) m = fmaxf(m, sc[k]);
    red[t] = m; __syncthreads();
    for (int s = 128; s > 0; s >>= 1) { if (t < s) red[t] = fmaxf(red[t], red[t + s]); __syncthreads(); }
    m = red[0]; __syncthreads();
    float ssum = 0.f;
    for (int k = t; k < Lk; k += 256) { float e = __expf(sc[k] - m); sc[k] = e; ssum += e; }
    red[t] = ssum; __syncthreads();
    for (int s = 128; s > 0; s >>= 1) { if (t < s) red[t] += red[t + s]; __syncthreads(); }
    float inv = 1.f / red[0];
    __syncthreads();
    if (t < 64) {
        float acc = 0.f;
        for (int k = 0; k < Lk; ++k)
            acc += sc[k] * vp[((size_t)(b * Lk + k)) * 512 + h * 64 + t];
        upd[((size_t)bh * u + i) * 64 + t] = acc * inv;
    }
}

// ctx = broadcast mean(V) (non-causal).  Thread per (b,h,d).
__global__ void k_ctxmean(const float* __restrict__ vp, float* __restrict__ ctx,
                          int Lq, int Lk) {
    int id = blockIdx.x * 256 + threadIdx.x; if (id >= 4096) return;
    int d = id & 63; int bh = id >> 6; int b = bh >> 3; int h = bh & 7;
    float s = 0.f;
    for (int k = 0; k < Lk; ++k) s += vp[((size_t)(b * Lk + k)) * 512 + h * 64 + d];
    s /= (float)Lk;
    float* c = ctx + (size_t)bh * Lq * 64 + d;
    for (int l = 0; l < Lq; ++l) c[(size_t)l * 64] = s;
}

// ctx = cumsum(V) (causal).  Thread per (b,h,d).
__global__ void k_cumsum(const float* __restrict__ vp, float* __restrict__ ctx, int L) {
    int id = blockIdx.x * 256 + threadIdx.x; if (id >= 4096) return;
    int d = id & 63; int bh = id >> 6; int b = bh >> 3; int h = bh & 7;
    float s = 0.f;
    float* c = ctx + (size_t)bh * L * 64 + d;
    for (int l = 0; l < L; ++l) {
        s += vp[((size_t)(b * L + l)) * 512 + h * 64 + d];
        c[(size_t)l * 64] = s;
    }
}

__global__ void k_scatter(float* __restrict__ ctx, const float* __restrict__ upd,
                          const int* __restrict__ idx, int L, int u) {
    int id = blockIdx.x * 256 + threadIdx.x;
    if (id >= 64 * u * 64) return;
    int d = id & 63; int rest = id >> 6; int i = rest % u; int bh = rest / u;
    int q = idx[bh * u + i];
    ctx[((size_t)bh * L + q) * 64 + d] = upd[((size_t)(bh * u + i)) * 64 + d];
}

// [b,h,l,d] f32 ctx -> [b*l, 512] f16 rows (for output projection GEMM)
__global__ void k_ctx_rows(const float* __restrict__ ctx, _Float16* __restrict__ dst, int L) {
    size_t id = (size_t)blockIdx.x * 256 + threadIdx.x;
    if (id >= (size_t)8 * L * 512) return;
    int c = (int)(id % 512); size_t row = id / 512;
    int l = (int)(row % L); int b = (int)(row / L);
    int h = c >> 6, d = c & 63;
    dst[id] = (_Float16)ctx[(((size_t)(b * 8 + h)) * L + l) * 64 + d];
}

// ---------------- distil: BN-ish scale + ELU, maxpool(3, stride2, pad -inf) -
__global__ void k_bn_elu(const float* __restrict__ x, const float* __restrict__ g,
                         const float* __restrict__ b, float* __restrict__ y, int n) {
    int id = blockIdx.x * 256 + threadIdx.x; if (id >= n) return;
    int c = id % 512;
    float v = x[id] * 0.9999950000374997f * g[c] + b[c];   // rsqrt(1+1e-5)
    y[id] = (v > 0.f) ? v : expm1f(v);
}

__global__ void k_maxpool(const float* __restrict__ y, float* __restrict__ out) {
    int id = blockIdx.x * 256 + threadIdx.x; if (id >= 2880 * 512) return;
    int c = id % 512; int r = id / 512; int lo = r % 360; int b = r / 360;
    float m = NEG_INF;
#pragma unroll
    for (int j = -1; j <= 1; ++j) {
        int l = 2 * lo + j;
        if (l >= 0 && l < 720) m = fmaxf(m, y[((size_t)(b * 720 + l)) * 512 + c]);
    }
    out[id] = m;
}

// ---------------- final projection + RevIN inverse + slice ------------------
__global__ void k_final(const float* __restrict__ dec, const float* __restrict__ pw,
    const float* __restrict__ pb, const float* __restrict__ rw,
    const float* __restrict__ rb, const float* __restrict__ mean,
    const float* __restrict__ stdv, float* __restrict__ out)
{
    int id = blockIdx.x * 256 + threadIdx.x;
    if (id >= 8 * 168 * 7) return;
    int c = id % 7; int r = id / 7; int l = r % 168; int b = r / 168;
    const float* drow = dec + (size_t)(b * 336 + 168 + l) * 512;
    float o = pb[c];
    for (int k = 0; k < 512; ++k) o += drow[k] * pw[k * 7 + c];
    o = (o - rb[c]) / (rw[c] + 1e-10f);
    out[id] = o * stdv[b * 7 + c] + mean[b * 7 + c];
}

// ---------------------------------------------------------------------------
extern "C" void kernel_launch(void* const* d_in, const int* in_sizes, int n_in,
                              void* d_out, int out_size, void* d_ws, size_t ws_size,
                              hipStream_t stream)
{
    (void)in_sizes; (void)n_in; (void)out_size; (void)ws_size;
    const float* x_enc = (const float*)d_in[0];
    const float* x_dec = (const float*)d_in[2];
    // params flattened in jax tree order (dict keys sorted), base index 4
    auto P = [&](int i) { return (const float*)d_in[4 + i]; };
    // rel indices:
    // dec0: b1=0 b2=1 ca{bk=2 bo=3 bq=4 bv=5 wk=6 wo=7 wq=8 wv=9}
    //       ln1_b=10 ln1_g=11 ln2_b=12 ln2_g=13 ln3_b=14 ln3_g=15
    //       sa{bk=16..wv=23} w1=24 w2=25
    // dec_conv=26 dec_ln_b=27 dec_ln_g=28
    // distil0: b=29 bn_b=30 bn_g=31 w=32
    // enc0: attn{bk=33..wv=40} b1=41 b2=42 ln1_b=43 ln1_g=44 ln2_b=45 ln2_g=46 w1=47 w2=48
    // enc1: attn{bk=49..wv=56} b1=57 b2=58 ln1_b=59 ln1_g=60 ln2_b=61 ln2_g=62 w1=63 w2=64
    // enc_conv=65 enc_ln_b=66 enc_ln_g=67 proj_b=68 proj_w=69 rev_b=70 rev_w=71

    struct AP { const float *wq,*bq,*wk,*bk,*wv,*bv,*wo,*bo; };
    auto mkAP = [&](int Ba) {
        AP a; a.bk=P(Ba); a.bo=P(Ba+1); a.bq=P(Ba+2); a.bv=P(Ba+3);
        a.wk=P(Ba+4); a.wo=P(Ba+5); a.wq=P(Ba+6); a.wv=P(Ba+7); return a;
    };
    AP enc0A = mkAP(33), enc1A = mkAP(49), saA = mkAP(16), caA = mkAP(2);

    // ---- workspace bump allocator (256B aligned offsets) ----
    char* wp = (char*)d_ws;
    auto alloc = [&](size_t bytes) -> void* {
        void* r = wp; wp += (bytes + 255) & ~(size_t)255; return r;
    };
    float* MEAN = (float*)alloc(56 * 4);
    float* STD  = (float*)alloc(56 * 4);
    float* XN   = (float*)alloc((size_t)8 * 720 * 7 * 4);
    float* X    = (float*)alloc((size_t)5760 * 512 * 4);   // encoder stream
    float* T1   = (float*)alloc((size_t)5760 * 512 * 4);
    float* T2   = (float*)alloc((size_t)5760 * 512 * 4);
    float* QP   = (float*)alloc((size_t)5760 * 512 * 4);
    float* KP   = (float*)alloc((size_t)5760 * 512 * 4);
    float* VP   = (float*)alloc((size_t)5760 * 512 * 4);
    float* CTX  = (float*)alloc((size_t)5760 * 512 * 4);
    float* FF   = (float*)alloc((size_t)5760 * 2048 * 4);
    float* DX   = (float*)alloc((size_t)2688 * 512 * 4);   // decoder stream
    float* MVAL = (float*)alloc((size_t)8 * 8 * 720 * 4);
    int*   IDX  = (int*)alloc((size_t)64 * 64 * 4);
    float* UPD  = (float*)alloc((size_t)64 * 64 * 64 * 4);
    _Float16* HX   = (_Float16*)alloc((size_t)5760 * 2048 * 2);  // f16 activations
    _Float16* WBUF = (_Float16*)alloc((size_t)2048 * 512 * 2);   // JIT f16 weights

    auto gemm = [&](const _Float16* A, const _Float16* Bt, const float* bias,
                    const float* res, float* C, int M, int N, int K, int act) {
        dim3 g((M + 255) / 256, N / 64, 1);
        k_gemm<<<g, 256, 0, stream>>>(A, Bt, bias, res, C, M, N, K, act);
    };
    auto prepw = [&](const float* src, int N, int K, int sn, int sk, int off) {
        int n = N * K;
        k_conv_w<<<(n + 255) / 256, 256, 0, stream>>>(src, WBUF, N, K, sn, sk, off);
    };
    auto f2h = [&](const float* src, _Float16* dst, size_t n) {
        k_f2h<<<(unsigned)((n + 255) / 256), 256, 0, stream>>>(src, dst, n);
    };

    auto run_attn = [&](const float* xq, int Lq, const float* xkv, int Lk,
                        const AP& p, bool causal, unsigned seed, float* outb) {
        int Mq = 8 * Lq, Mk = 8 * Lk;
        f2h(xq, HX, (size_t)Mq * 512);
        prepw(p.wq, 512, 512, 1, 512, 0);
        gemm(HX, WBUF, p.bq, nullptr, QP, Mq, 512, 512, 0);
        f2h(xkv, HX, (size_t)Mk * 512);
        prepw(p.wk, 512, 512, 1, 512, 0);
        gemm(HX, WBUF, p.bk, nullptr, KP, Mk, 512, 512, 0);
        prepw(p.wv, 512, 512, 1, 512, 0);
        gemm(HX, WBUF, p.bv, nullptr, VP, Mk, 512, 512, 0);
        int u  = 5 * (int)std::ceil(std::log((double)Lq)); if (u  > Lq) u  = Lq;
        int Up = 5 * (int)std::ceil(std::log((double)Lk)); if (Up > Lk) Up = Lk;
        int waves = 64 * Lq;
        k_sample<<<(waves * 32 + 255) / 256, 256, 0, stream>>>(QP, KP, MVAL, Lq, Lk, Up, seed);
        k_topk<<<64, 256, 0, stream>>>(MVAL, IDX, Lq, u);
        k_attn_sel<<<64 * u, 256, 0, stream>>>(QP, KP, VP, IDX, UPD, Lq, Lk, u, causal ? 1 : 0);
        if (causal) k_cumsum<<<16, 256, 0, stream>>>(VP, CTX, Lq);
        else        k_ctxmean<<<16, 256, 0, stream>>>(VP, CTX, Lq, Lk);
        k_scatter<<<(64 * u * 64 + 255) / 256, 256, 0, stream>>>(CTX, UPD, IDX, Lq, u);
        k_ctx_rows<<<(Mq * 512 + 255) / 256, 256, 0, stream>>>(CTX, HX, Lq);
        prepw(p.wo, 512, 512, 1, 512, 0);
        gemm(HX, WBUF, p.bo, nullptr, outb, Mq, 512, 512, 0);
    };

    auto attn_block = [&](float* x, int Lq, const float* xkv, int Lk, const AP& p,
                          bool causal, unsigned seed, const float* lng, const float* lnb) {
        int M = 8 * Lq;
        run_attn(x, Lq, xkv, Lk, p, causal, seed, T2);
        k_add<<<(M * 512 + 255) / 256, 256, 0, stream>>>(x, T2, T2, M * 512);
        k_ln<<<M, 256, 0, stream>>>(T2, nullptr, lng, lnb, x);
    };

    auto ffn = [&](float* x, int M, const float* w1, const float* b1,
                   const float* w2, const float* b2, const float* lng, const float* lnb) {
        f2h(x, HX, (size_t)M * 512);
        prepw(w1, 2048, 512, 1, 2048, 0);
        gemm(HX, WBUF, b1, nullptr, FF, M, 2048, 512, 1);        // GELU fused
        f2h(FF, HX, (size_t)M * 2048);
        prepw(w2, 512, 2048, 1, 512, 0);
        gemm(HX, WBUF, b2, nullptr, T1, M, 512, 2048, 0);
        k_ln<<<M, 256, 0, stream>>>(x, T1, lng, lnb, x);          // LN(x + y)
    };

    // ================= pipeline =================
    // RevIN
    k_revin_stats<<<1, 64, 0, stream>>>(x_enc, MEAN, STD);
    k_revin_norm<<<(8 * 720 * 7 + 255) / 256, 256, 0, stream>>>(x_enc, MEAN, STD, P(71), P(70), XN);

    // encoder embedding (circular conv 7->512 + positional encoding)
    k_embed<<<(8 * 720 * 512 + 255) / 256, 256, 0, stream>>>(XN, P(65), X, 720);

    // encoder layer 0 (L=720)
    attn_block(X, 720, X, 720, enc0A, false, 0x1111u, P(44), P(43));
    ffn(X, 5760, P(47), P(41), P(48), P(42), P(46), P(45));

    // distil: circular conv(512->512,k=3) as 3 accumulated WMMA GEMMs
    for (int t = 0; t < 3; ++t) {
        k_shift_f16<<<(5760 * 512 + 255) / 256, 256, 0, stream>>>(X, HX, 720, t - 1);
        prepw(P(32), 512, 512, 1536, 3, t);                      // w[o][i][t] slice
        gemm(HX, WBUF, (t == 2) ? P(29) : nullptr,
             (t == 0) ? nullptr : T1, T1, 5760, 512, 512, 0);
    }
    k_bn_elu<<<(5760 * 512 + 255) / 256, 256, 0, stream>>>(T1, P(31), P(30), T1, 5760 * 512);
    k_maxpool<<<(2880 * 512 + 255) / 256, 256, 0, stream>>>(T1, X);   // X: [B,360,512]

    // encoder layer 1 (L=360) + final encoder LN
    attn_block(X, 360, X, 360, enc1A, false, 0x2222u, P(60), P(59));
    ffn(X, 2880, P(63), P(57), P(64), P(58), P(62), P(61));
    k_ln<<<2880, 256, 0, stream>>>(X, nullptr, P(67), P(66), X);

    // decoder embedding (raw x_dec)
    k_embed<<<(8 * 336 * 512 + 255) / 256, 256, 0, stream>>>(x_dec, P(26), DX, 336);

    // decoder layer: causal self-attn, cross-attn vs encoder, FFN
    attn_block(DX, 336, DX, 336, saA, true, 0x3333u, P(11), P(10));
    attn_block(DX, 336, X, 360, caA, false, 0x4444u, P(13), P(12));
    ffn(DX, 2688, P(24), P(0), P(25), P(1), P(15), P(14));
    k_ln<<<2688, 256, 0, stream>>>(DX, nullptr, P(28), P(27), DX);

    // projection to 7 channels + inverse RevIN + take last 168 steps
    k_final<<<(8 * 168 * 7 + 255) / 256, 256, 0, stream>>>(
        DX, P(69), P(68), P(71), P(70), MEAN, STD, (float*)d_out);
}
